// BlockLocalAttentionProduct_20564303413308
// MI455X (gfx1250) — compile-verified, matrix operand order checked by reading, not run on hardware
//
#include <hip/hip_runtime.h>
#include <cstdint>
#include <cstddef>

typedef __attribute__((ext_vector_type(16))) _Float16 v16h;
typedef __attribute__((ext_vector_type(8)))  _Float16 v8h;
typedef __attribute__((ext_vector_type(8)))  float    v8f;

#define S_LEN   4096
#define NHEAD   12
#define DHEAD   64
#define CHUNKW  256
#define HALF_W  128
#define NWAVES  8
#define NTHREADS (NWAVES * 32)

// LDS strides in halfs: multiples of 8 (16B alignment for b128) and picked so
// per-row dword offsets (36 / 132 / 20) are coprime-ish with 64 banks -> no conflicts.
#define K_STRIDE   72      // 64 + 8 pad  (144 B/row)
#define VT_STRIDE  264     // 256 + 8 pad (528 B/row)
#define P_STRIDE   40      // 32 + 8 pad  (80 B/row)
#define LDS_BYTES  ((CHUNKW * K_STRIDE + DHEAD * VT_STRIDE + NWAVES * 16 * P_STRIDE) * 2)

// Fragment layout (16-bit A 16x32, symmetric B): lane holds row (A) / column (B),
// 8 contiguous halfs at k0 + 8*(lane/16) and 8 more at +16 halfs.
__device__ inline v16h ld_frag(const _Float16* p) {
  v8h lo = *(const v8h*)(p);
  v8h hi = *(const v8h*)(p + 16);
  v16h f;
#pragma unroll
  for (int i = 0; i < 8; ++i) { f[i] = lo[i]; f[i + 8] = hi[i]; }
  return f;
}

__device__ inline v16h make_frag(float4 a0, float4 a1, float4 b0, float4 b1) {
  v16h f;
  f[0]  = (_Float16)a0.x; f[1]  = (_Float16)a0.y; f[2]  = (_Float16)a0.z; f[3]  = (_Float16)a0.w;
  f[4]  = (_Float16)a1.x; f[5]  = (_Float16)a1.y; f[6]  = (_Float16)a1.z; f[7]  = (_Float16)a1.w;
  f[8]  = (_Float16)b0.x; f[9]  = (_Float16)b0.y; f[10] = (_Float16)b0.z; f[11] = (_Float16)b0.w;
  f[12] = (_Float16)b1.x; f[13] = (_Float16)b1.y; f[14] = (_Float16)b1.z; f[15] = (_Float16)b1.w;
  return f;
}

__global__ __launch_bounds__(NTHREADS)
void block_attn_win_kernel(const float* __restrict__ Q, const float* __restrict__ K,
                           const float* __restrict__ V, const float* __restrict__ M,
                           float* __restrict__ out, int red) {
  extern __shared__ _Float16 smem[];
  _Float16* Kh = smem;                               // [256][K_STRIDE]  f16 K rows
  _Float16* Vt = Kh + CHUNKW * K_STRIDE;             // [64][VT_STRIDE]  f16 V^T rows
  _Float16* Pb = Vt + DHEAD * VT_STRIDE;             // [NWAVES][16][P_STRIDE] P transpose scratch

  const int win  = blockIdx.x;
  const int bh   = blockIdx.y;                       // b*H + h
  const int b    = bh / NHEAD;
  const int wbase = red ? (HALF_W + win * CHUNKW) : (win * CHUNKW);

  const int tid  = threadIdx.x;
  const int wave = tid >> 5;
  const int lane = tid & 31;
  const int lm   = lane & 15;
  const int hh   = lane >> 4;

  const float* Qb = Q + ((size_t)bh * S_LEN + wbase) * DHEAD;
  const float* Kb = K + ((size_t)bh * S_LEN + wbase) * DHEAD;
  const float* Vb = V + ((size_t)bh * S_LEN + wbase) * DHEAD;
  float*       Ob = out + ((size_t)bh * S_LEN + wbase) * DHEAD;
  const float* Mb = M + (size_t)b * S_LEN + wbase;

  __builtin_prefetch(Qb + (size_t)tid * DHEAD, 0, 0);   // global_prefetch_b8

  // ---- stage K (row-major f16) and V^T (f16) into LDS; 1 thread per seq row ----
  {
    const int row = tid;
    const float* kr = Kb + (size_t)row * DHEAD;
    const float* vr = Vb + (size_t)row * DHEAD;
#pragma unroll
    for (int c = 0; c < DHEAD; c += 8) {
      float4 x = *(const float4*)(kr + c);
      float4 y = *(const float4*)(kr + c + 4);
      v8h hv = { (_Float16)x.x, (_Float16)x.y, (_Float16)x.z, (_Float16)x.w,
                 (_Float16)y.x, (_Float16)y.y, (_Float16)y.z, (_Float16)y.w };
      *(v8h*)(Kh + row * K_STRIDE + c) = hv;
    }
#pragma unroll
    for (int c = 0; c < DHEAD; c += 4) {
      float4 x = *(const float4*)(vr + c);
      Vt[(c + 0) * VT_STRIDE + row] = (_Float16)x.x;
      Vt[(c + 1) * VT_STRIDE + row] = (_Float16)x.y;
      Vt[(c + 2) * VT_STRIDE + row] = (_Float16)x.z;
      Vt[(c + 3) * VT_STRIDE + row] = (_Float16)x.w;
    }
  }
  __syncthreads();

  // mask value per key-column tile (mask broadcast over query rows)
  float mval[16];
#pragma unroll
  for (int t = 0; t < 16; ++t) mval[t] = Mb[16 * t + lm];

  _Float16* Pw = Pb + wave * 16 * P_STRIDE;
  const v8f zero8 = {0.f, 0.f, 0.f, 0.f, 0.f, 0.f, 0.f, 0.f};

  for (int it = 0; it < 2; ++it) {
    const int strip = wave + NWAVES * it;     // 0..15
    const int rbase = strip * 16;             // query-row base within window

    // ---- Q A-fragments (k0 = 0 and 32), direct from global with f32->f16 cvt ----
    const float* qr = Qb + (size_t)(rbase + lm) * DHEAD + 8 * hh;
    v16h aq0 = make_frag(*(const float4*)(qr +  0), *(const float4*)(qr +  4),
                         *(const float4*)(qr + 16), *(const float4*)(qr + 20));
    v16h aq1 = make_frag(*(const float4*)(qr + 32), *(const float4*)(qr + 36),
                         *(const float4*)(qr + 48), *(const float4*)(qr + 52));

    // ---- scores: S = Q * K^T  (16x256 strip, 16 tiles, 2 WMMAs each) ----
    v8f acc[16];
#pragma unroll
    for (int t = 0; t < 16; ++t) {
      const _Float16* kp = Kh + (16 * t + lm) * K_STRIDE + 8 * hh;
      v16h bk0 = ld_frag(kp);
      v16h bk1 = ld_frag(kp + 32);
      v8f c = zero8;
      c = __builtin_amdgcn_wmma_f32_16x16x32_f16(false, aq0, false, bk0, (short)0, c, false, false);
      c = __builtin_amdgcn_wmma_f32_16x16x32_f16(false, aq1, false, bk1, (short)0, c, false, false);
      acc[t] = c;
    }

    // ---- softmax over 256 keys, per row (8 rows per lane-half) ----
    float inv[8];
#pragma unroll
    for (int r = 0; r < 8; ++r) {
      float m = -3.0e38f;
#pragma unroll
      for (int t = 0; t < 16; ++t) {
        float x = acc[t][r] * 0.125f + mval[t];   // 1/sqrt(64) + mask
        acc[t][r] = x;
        m = fmaxf(m, x);
      }
#pragma unroll
      for (int off = 1; off < 16; off <<= 1) m = fmaxf(m, __shfl_xor(m, off));
      float s = 0.f;
#pragma unroll
      for (int t = 0; t < 16; ++t) {
        float e = __expf(acc[t][r] - m);
        acc[t][r] = e;
        s += e;
      }
#pragma unroll
      for (int off = 1; off < 16; off <<= 1) s += __shfl_xor(s, off);
      inv[r] = 1.0f / s;
    }

    // ---- ctx = P * V : transpose P 16x32-chunkwise through wave-private LDS ----
    v8f co[4] = {zero8, zero8, zero8, zero8};
#pragma unroll
    for (int j = 0; j < 8; ++j) {
#pragma unroll
      for (int tt = 0; tt < 2; ++tt) {
        const int t = 2 * j + tt;
#pragma unroll
        for (int r = 0; r < 8; ++r) {
          Pw[(r + 8 * hh) * P_STRIDE + tt * 16 + lm] = (_Float16)(acc[t][r] * inv[r]);
        }
      }
      // wave-internal LDS ordering: loads below see the stores above
      v16h ap = ld_frag(Pw + lm * P_STRIDE + 8 * hh);
#pragma unroll
      for (int i = 0; i < 4; ++i) {
        const _Float16* vp = Vt + (16 * i + lm) * VT_STRIDE + 32 * j + 8 * hh;
        v16h bv = ld_frag(vp);
        co[i] = __builtin_amdgcn_wmma_f32_16x16x32_f16(false, ap, false, bv, (short)0, co[i], false, false);
      }
    }

    // ---- weighted write-out ----
    const int growb = wbase + rbase;   // absolute sequence row base of strip
    const float wgt = red ? 0.5f
                          : ((growb < HALF_W || growb >= S_LEN - HALF_W) ? 1.0f : 0.5f);
#pragma unroll
    for (int i = 0; i < 4; ++i) {
#pragma unroll
      for (int r = 0; r < 8; ++r) {
        float* o = Ob + (size_t)(rbase + r + 8 * hh) * DHEAD + 16 * i + lm;
        const float v = co[i][r] * wgt;
        if (red) *o += v; else *o = v;
      }
    }
  }
}

extern "C" void kernel_launch(void* const* d_in, const int* in_sizes, int n_in,
                              void* d_out, int out_size, void* d_ws, size_t ws_size,
                              hipStream_t stream) {
  (void)in_sizes; (void)n_in; (void)out_size; (void)d_ws; (void)ws_size;
  const float* Q = (const float*)d_in[0];
  const float* K = (const float*)d_in[1];
  const float* V = (const float*)d_in[2];
  const float* M = (const float*)d_in[3];
  float* out = (float*)d_out;

  dim3 blk(NTHREADS);
  dim3 gridFull(S_LEN / CHUNKW, 4 * NHEAD);        // 16 x 48 : writes every output row
  dim3 gridRed (S_LEN / CHUNKW - 1, 4 * NHEAD);    // 15 x 48 : accumulates 0.5*red

  block_attn_win_kernel<<<gridFull, blk, LDS_BYTES, stream>>>(Q, K, V, M, out, 0);
  block_attn_win_kernel<<<gridRed,  blk, LDS_BYTES, stream>>>(Q, K, V, M, out, 1);
}
